// NeuralMemory_32435593019747
// MI455X (gfx1250) — matrix operand order repaired and checked by reading, not compiled, not run on hardware
//
#include <hip/hip_runtime.h>
#include <math.h>

typedef __attribute__((ext_vector_type(16))) _Float16 v16h;
typedef __attribute__((ext_vector_type(8)))  float    v8f;

#define NB 8
#define NT 512
#define NE 64
#define NH 128

// ---------------- workspace layout (float element offsets) ----------------
#define WS_KN    0                          // B*T*E   normalized keys (t-row-major)
#define WS_VN    (WS_KN + NB*NT*NE)         // B*T*E   normalized values (t-row-major)
#define WS_KT    (WS_VN + NB*NT*NE)         // B*E*T   keys transposed (e-major)
#define WS_HT    (WS_KT + NB*NE*NT)         // B*H*T   hidden silu(Z) transposed (h-major)
#define WS_DZT   (WS_HT + NB*NH*NT)         // B*H*T   dL/dZ transposed
#define WS_DYT   (WS_DZT + NB*NH*NT)        // B*E*T   dL/dY transposed
#define WS_NK    (WS_DYT + NB*NE*NT)        // B*E     key column sumsq (over T)
#define WS_NV    (WS_NK + NB*NE)            // B*E     value column sumsq
#define WS_BC    (WS_NV + NB*NE)            // T       B_coeffs
#define WS_DC    (WS_BC + NT)               // T       D_coeffs
#define WS_SC    (WS_DC + NT)               // 4       {p_T, q_T, A_T}
// total ~2.10M floats (~8.4 MB)

// ---------------- output layout (floats) ----------------
#define O_LOSS 0
#define O_MW1  512
#define O_MB1  (O_MW1 + NB*NH*NE)   // 66048
#define O_MW2  (O_MB1 + NB*NH)      // 67072
#define O_MB2  (O_MW2 + NB*NE*NH)   // 132608
#define O_SW1  (O_MB2 + NB*NE)      // 133120
#define O_SB1  (O_SW1 + NB*NH*NE)   // 198656
#define O_SW2  (O_SB1 + NB*NH)      // 199680
#define O_SB2  (O_SW2 + NB*NE*NH)   // 265216

// fast sigmoid: v_exp_f32 + v_add_f32 + v_rcp_f32 (TRANS ops co-execute with WMMA)
__device__ __forceinline__ float fast_sigmoid(float z) {
  return __builtin_amdgcn_rcpf(1.f + __expf(-z));
}

// ---------------- WMMA fragment helpers (wave32, 16x16x32 f16->f32) ----------------
__device__ __forceinline__ v16h cvt16(float4 a, float4 b, float4 c, float4 d) {
  v16h r;
  r[0]=(_Float16)a.x;  r[1]=(_Float16)a.y;  r[2]=(_Float16)a.z;  r[3]=(_Float16)a.w;
  r[4]=(_Float16)b.x;  r[5]=(_Float16)b.y;  r[6]=(_Float16)b.z;  r[7]=(_Float16)b.w;
  r[8]=(_Float16)c.x;  r[9]=(_Float16)c.y;  r[10]=(_Float16)c.z; r[11]=(_Float16)c.w;
  r[12]=(_Float16)d.x; r[13]=(_Float16)d.y; r[14]=(_Float16)d.z; r[15]=(_Float16)d.w;
  return r;
}

// A 16x32 f16 fragment: lane&15 = row m; lane-half selects contiguous K block of 16.
// Row-major source -> 16 contiguous floats (4x b128).
__device__ __forceinline__ v16h load_a(const float* src, int ld, int m, int kk, int half) {
  const float*  p  = src + m * ld + kk + half * 16;
  const float4* q  = (const float4*)p;
  return cvt16(q[0], q[1], q[2], q[3]);
}

// Same, with per-K coefficient scale (coeff indexed identically to K)
__device__ __forceinline__ v16h load_a_scaled(const float* src, int ld, int m, int kk,
                                              int half, const float* coeff) {
  const float*  p   = src + m * ld + kk + half * 16;
  const float*  cp  = coeff + kk + half * 16;
  const float4* q   = (const float4*)p;
  const float4* cq  = (const float4*)cp;
  float4 a = q[0], b = q[1], c = q[2], d = q[3];
  float4 ca = cq[0], cb = cq[1], cc = cq[2], cd = cq[3];
  a.x *= ca.x; a.y *= ca.y; a.z *= ca.z; a.w *= ca.w;
  b.x *= cb.x; b.y *= cb.y; b.z *= cb.z; b.w *= cb.w;
  c.x *= cc.x; c.y *= cc.y; c.z *= cc.z; c.w *= cc.w;
  d.x *= cd.x; d.y *= cd.y; d.z *= cd.z; d.w *= cd.w;
  return cvt16(a, b, c, d);
}

// B 32x16 f16 fragment: lane&15 = column n; lane-half selects contiguous K block of 16.
// Source stored as NxK row-major (transposed view) -> 16 contiguous floats (4x b128).
__device__ __forceinline__ v16h load_bT(const float* src, int ld, int kk, int half, int n) {
  const float4* q = (const float4*)(src + n * ld + kk + half * 16);
  return cvt16(q[0], q[1], q[2], q[3]);
}

// B from row-major KxN source (strided columns; scalar loads)
__device__ __forceinline__ v16h load_b(const float* src, int ld, int kk, int half, int n) {
  v16h bm;
#pragma unroll
  for (int j = 0; j < 16; ++j) bm[j] = (_Float16)src[(kk + half * 16 + j) * ld + n];
  return bm;
}

__device__ __forceinline__ v8f wmma_f(v16h a, v16h b, v8f c) {
  return __builtin_amdgcn_wmma_f32_16x16x32_f16(false, a, false, b, (short)0, c, false, false);
}

// ---------------- K0: zero reductions + sequential coefficient recurrences ----------------
__global__ void init_coeffs_kernel(float* ws, float* out) {
  int tid = threadIdx.x;
  for (int i = tid; i < NB * NE; i += blockDim.x) { ws[WS_NK + i] = 0.f; ws[WS_NV + i] = 0.f; }
  for (int i = tid; i < NT; i += blockDim.x) out[O_LOSS + i] = 0.f;
  if (tid == 0) {
    const double beta = 1.0 - 0.999, eta = 0.95, theta = 0.05;
    double qp[NT];                       // eta^(t+1)
    double acc = 1.0;
    for (int t = 0; t < NT; ++t) { acc *= eta; qp[t] = acc; }
    double suffix = 0.0, bpow = 1.0, dpow = eta;   // bpow = beta^(T-1-t), dpow = eta^(T-t)
    for (int t = NT - 1; t >= 0; --t) {
      double w = bpow * qp[t];
      suffix += w;
      ws[WS_BC + t] = (float)(-theta * suffix / qp[t]);
      ws[WS_DC + t] = (float)(-theta * dpow);
      bpow *= beta; dpow *= eta;
    }
    ws[WS_SC + 0] = (float)bpow;         // p_T
    ws[WS_SC + 1] = (float)qp[NT - 1];   // q_T
    ws[WS_SC + 2] = (float)suffix;       // A_T
  }
}

// ---------------- K1: silu(x @ W^T) + column sumsq (one wave per 16xE strip) ----------------
__global__ void kv_kernel(const float* __restrict__ x, const float* __restrict__ WK,
                          const float* __restrict__ WV, float* ws) {
  int lane = threadIdx.x;
  int m = lane & 15, half = lane >> 4;
  int t0 = blockIdx.x * 16;
  int b = blockIdx.y;
  int which = blockIdx.z;
  const float* W  = which ? WV : WK;
  float* dst = ws + (which ? WS_VN : WS_KN) + (size_t)(b * NT + t0) * NE;
  float* nrm = ws + (which ? WS_NV : WS_NK) + b * NE;
  const float* xa = x + (size_t)(b * NT + t0) * NE;
#pragma unroll
  for (int nt = 0; nt < 4; ++nt) {
    int n0 = nt * 16;
    v8f accv = {};
#pragma unroll
    for (int kk = 0; kk < NE; kk += 32) {
      v16h a  = load_a(xa, NE, m, kk, half);
      v16h bm = load_bT(W, NE, kk, half, n0 + m);   // B[k,e] = W[e,k]
      accv = wmma_f(a, bm, accv);
    }
    float ss = 0.f;
#pragma unroll
    for (int r = 0; r < 8; ++r) {
      float z = accv[r];
      float s = z * fast_sigmoid(z);                // silu
      dst[(half * 8 + r) * NE + n0 + m] = s;
      ss += s * s;
    }
    atomicAdd(&nrm[n0 + m], ss);
  }
}

// ---------------- K2: divide by max(||col||,1e-12); emit e-major key copy ----------------
__global__ void normalize_kernel(float* ws) {
  int idx = blockIdx.x * blockDim.x + threadIdx.x;
  if (idx >= 2 * NB * NT * NE) return;
  int which = idx / (NB * NT * NE);
  int r = idx % (NB * NT * NE);
  int b = r / (NT * NE);
  int t = (r / NE) % NT;
  int e = r % NE;
  float* buf = ws + (which ? WS_VN : WS_KN);
  const float* nrm = ws + (which ? WS_NV : WS_NK);
  float n = sqrtf(nrm[b * NE + e]);
  float v = buf[r] * __builtin_amdgcn_rcpf(fmaxf(n, 1e-12f));
  buf[r] = v;
  if (!which) ws[WS_KT + ((size_t)b * NE + e) * NT + t] = v;   // transposed keys for grad GEMM
}

// ---------------- K3: fused fwd+bwd over a 16-token strip (48 WMMAs/block) ----------------
__global__ void forward_kernel(const float* __restrict__ W1, const float* __restrict__ b1,
                               const float* __restrict__ W2, const float* __restrict__ b2,
                               float* ws, float* out) {
  __shared__ __align__(16) float Hs_l[16 * NH];
  __shared__ __align__(16) float Sg_l[16 * NH];
  __shared__ __align__(16) float Dy_l[16 * NE];
  __shared__ __align__(16) float Vl[16 * NE];
  __shared__ float loss_l[16];
  int lane = threadIdx.x;
  int m = lane & 15, half = lane >> 4;
  int t0 = blockIdx.x * 16;
  int b = blockIdx.y;
  const float* Kn = ws + WS_KN + (size_t)(b * NT + t0) * NE;
  const float* Vn = ws + WS_VN + (size_t)(b * NT + t0) * NE;
  float* HTb  = ws + WS_HT  + (size_t)b * NH * NT;
  float* dZTb = ws + WS_DZT + (size_t)b * NH * NT;
  float* dYTb = ws + WS_DYT + (size_t)b * NE * NT;
  const float* W1b = W1 + (size_t)b * NH * NE;
  const float* W2b = W2 + (size_t)b * NE * NH;

  // gfx1250 async DMA: stage the 16xE value strip into LDS while stage-1 GEMM runs.
  {
    unsigned lbase = (unsigned)(size_t)(&Vl[0]);
    const char* gbase = (const char*)Vn;
#pragma unroll
    for (int i = 0; i < 8; ++i) {
      unsigned off = (unsigned)((i * 32 + lane) * 16);  // 256 x b128 chunks = 16*64 floats
      unsigned laddr = lbase + off;
      const char* ga = gbase + off;
      asm volatile("global_load_async_to_lds_b128 %0, %1, off"
                   :: "v"(laddr), "v"(ga) : "memory");
    }
  }

  if (lane < 16) loss_l[lane] = 0.f;
  __syncthreads();

  // Z = Kn @ W1^T + b1 ; Hs = silu(Z) ; Sg = silu'(Z) ; HT (t-major) for grad GEMM
  for (int ht = 0; ht < 8; ++ht) {
    int h0 = ht * 16;
    v8f accv = {};
#pragma unroll
    for (int kk = 0; kk < NE; kk += 32) {
      v16h a  = load_a(Kn, NE, m, kk, half);
      v16h bm = load_bT(W1b, NE, kk, half, h0 + m);  // B[e,h] = W1[h,e]
      accv = wmma_f(a, bm, accv);
    }
    float bias = b1[b * NH + h0 + m];
    float tmp[8];
#pragma unroll
    for (int r = 0; r < 8; ++r) {
      float z  = accv[r] + bias;
      float sg = fast_sigmoid(z);
      float hs = z * sg;
      float sp = sg * (1.f + z * (1.f - sg));        // silu'
      int row = half * 8 + r;
      Hs_l[row * NH + h0 + m] = hs;
      Sg_l[row * NH + h0 + m] = sp;
      tmp[r] = hs;
    }
    float4* hp = (float4*)(HTb + (size_t)(h0 + m) * NT + t0 + half * 8);
    hp[0] = (float4){tmp[0], tmp[1], tmp[2], tmp[3]};
    hp[1] = (float4){tmp[4], tmp[5], tmp[6], tmp[7]};
  }
  asm volatile("s_wait_asynccnt 0" ::: "memory");
  __syncthreads();

  // Y = Hs @ W2^T + b2 ; R = Y - V ; dY = R/256 ; per-token loss partials
  for (int et = 0; et < 4; ++et) {
    int e0 = et * 16;
    v8f accv = {};
#pragma unroll
    for (int kk = 0; kk < NH; kk += 32) {
      v16h a  = load_a(Hs_l, NH, m, kk, half);
      v16h bm = load_bT(W2b, NH, kk, half, e0 + m);  // B[h,e] = W2[e,h]
      accv = wmma_f(a, bm, accv);
    }
    float bias = b2[b * NE + e0 + m];
    float tmp[8];
#pragma unroll
    for (int r = 0; r < 8; ++r) {
      int row = half * 8 + r;
      float resid = accv[r] + bias - Vl[row * NE + e0 + m];
      float dy = resid * (1.f / 256.f);              // 2/(B*E)
      Dy_l[row * NE + e0 + m] = dy;
      tmp[r] = dy;
      atomicAdd(&loss_l[row], resid * resid);
    }
    float4* dp = (float4*)(dYTb + (size_t)(e0 + m) * NT + t0 + half * 8);
    dp[0] = (float4){tmp[0], tmp[1], tmp[2], tmp[3]};
    dp[1] = (float4){tmp[4], tmp[5], tmp[6], tmp[7]};
  }
  __syncthreads();

  // dH = dY @ W2 ; dZ = dH * silu'(Z) -> dZT (t-major)
  for (int ht = 0; ht < 8; ++ht) {
    int h0 = ht * 16;
    v8f accv = {};
#pragma unroll
    for (int kk = 0; kk < NE; kk += 32) {
      v16h a  = load_a(Dy_l, NE, m, kk, half);
      v16h bm = load_b(W2b, NH, kk, half, h0 + m);   // B[e,h] = W2[e,h]
      accv = wmma_f(a, bm, accv);
    }
    float tmp[8];
#pragma unroll
    for (int r = 0; r < 8; ++r) {
      int row = half * 8 + r;
      tmp[r] = accv[r] * Sg_l[row * NH + h0 + m];
    }
    float4* zp = (float4*)(dZTb + (size_t)(h0 + m) * NT + t0 + half * 8);
    zp[0] = (float4){tmp[0], tmp[1], tmp[2], tmp[3]};
    zp[1] = (float4){tmp[4], tmp[5], tmp[6], tmp[7]};
  }
  __syncthreads();
  if (lane < 16) atomicAdd(&out[O_LOSS + t0 + lane], loss_l[lane] * (1.f / 512.f));
}

// ---------------- K4: M_W1/S_W1 tile = sum_t c[t]*dZ[t,h]*K[t,e] (K=T GEMM, 32 WMMAs) ----
__global__ void gradW1_kernel(const float* __restrict__ W1, const float* __restrict__ SW1,
                              const float* ws, float* out) {
  int lane = threadIdx.x;
  int m = lane & 15, half = lane >> 4;
  int m0 = blockIdx.x * 16;   // H
  int n0 = blockIdx.y * 16;   // E
  int b  = blockIdx.z;
  const float* dZT = ws + WS_DZT + (size_t)b * NH * NT;
  const float* KnT = ws + WS_KT  + (size_t)b * NE * NT;
  const float* Bc = ws + WS_BC;
  const float* Dc = ws + WS_DC;
  v8f accB = {}, accD = {};
  for (int kk = 0; kk < NT; kk += 32) {
    if (kk + 32 < NT) {
      __builtin_prefetch(dZT + (size_t)(m0 + m) * NT + kk + 32, 0, 0);
      __builtin_prefetch(KnT + (size_t)(n0 + m) * NT + kk + 32, 0, 0);
    }
    v16h bm = load_bT(KnT, NT, kk, half, n0 + m);                 // B[t,e] (contig t)
    v16h aB = load_a_scaled(dZT, NT, m0 + m, kk, half, Bc);       // A[h,t] = Bc[t]*dZ[t,h]
    v16h aD = load_a_scaled(dZT, NT, m0 + m, kk, half, Dc);
    accB = wmma_f(aB, bm, accB);
    accD = wmma_f(aD, bm, accD);
  }
  float pT = ws[WS_SC + 0], qT = ws[WS_SC + 1], AT = ws[WS_SC + 2];
#pragma unroll
  for (int r = 0; r < 8; ++r) {
    int h = m0 + half * 8 + r;
    int e = n0 + m;
    int idx = (b * NH + h) * NE + e;
    out[O_MW1 + idx] = pT * W1[idx] + AT * SW1[idx] + accB[r];
    out[O_SW1 + idx] = qT * SW1[idx] + accD[r];
  }
}

// ---------------- K5: M_W2/S_W2 tile = sum_t c[t]*dY[t,e]*Hs[t,h] ----------------
__global__ void gradW2_kernel(const float* __restrict__ W2, const float* __restrict__ SW2,
                              const float* ws, float* out) {
  int lane = threadIdx.x;
  int m = lane & 15, half = lane >> 4;
  int m0 = blockIdx.x * 16;   // E
  int n0 = blockIdx.y * 16;   // H
  int b  = blockIdx.z;
  const float* dYT = ws + WS_DYT + (size_t)b * NE * NT;
  const float* HT  = ws + WS_HT  + (size_t)b * NH * NT;
  const float* Bc = ws + WS_BC;
  const float* Dc = ws + WS_DC;
  v8f accB = {}, accD = {};
  for (int kk = 0; kk < NT; kk += 32) {
    if (kk + 32 < NT) {
      __builtin_prefetch(dYT + (size_t)(m0 + m) * NT + kk + 32, 0, 0);
      __builtin_prefetch(HT + (size_t)(n0 + m) * NT + kk + 32, 0, 0);
    }
    v16h bm = load_bT(HT, NT, kk, half, n0 + m);                  // B[t,h] (contig t)
    v16h aB = load_a_scaled(dYT, NT, m0 + m, kk, half, Bc);       // A[e,t] = Bc[t]*dY[t,e]
    v16h aD = load_a_scaled(dYT, NT, m0 + m, kk, half, Dc);
    accB = wmma_f(aB, bm, accB);
    accD = wmma_f(aD, bm, accD);
  }
  float pT = ws[WS_SC + 0], qT = ws[WS_SC + 1], AT = ws[WS_SC + 2];
#pragma unroll
  for (int r = 0; r < 8; ++r) {
    int e = m0 + half * 8 + r;
    int h = n0 + m;
    int idx = (b * NE + e) * NH + h;
    out[O_MW2 + idx] = pT * W2[idx] + AT * SW2[idx] + accB[r];
    out[O_SW2 + idx] = qT * SW2[idx] + accD[r];
  }
}

// ---------------- K6: bias-grad weighted sums + state pack (contiguous t reads) ----------
__global__ void bias_kernel(const float* __restrict__ b1, const float* __restrict__ Sb1,
                            const float* __restrict__ b2, const float* __restrict__ Sb2,
                            const float* ws, float* out) {
  int idx = blockIdx.x * blockDim.x + threadIdx.x;
  const float* Bc = ws + WS_BC;
  const float* Dc = ws + WS_DC;
  float pT = ws[WS_SC + 0], qT = ws[WS_SC + 1], AT = ws[WS_SC + 2];
  if (idx < NB * NH) {
    const float* g = ws + WS_DZT + (size_t)idx * NT;   // (b*NH + h)*NT
    float sB = 0.f, sD = 0.f;
    for (int t = 0; t < NT; ++t) { sB += Bc[t] * g[t]; sD += Dc[t] * g[t]; }
    out[O_MB1 + idx] = pT * b1[idx] + AT * Sb1[idx] + sB;
    out[O_SB1 + idx] = qT * Sb1[idx] + sD;
  } else if (idx < NB * NH + NB * NE) {
    int j = idx - NB * NH;
    const float* g = ws + WS_DYT + (size_t)j * NT;     // (b*NE + e)*NT
    float sB = 0.f, sD = 0.f;
    for (int t = 0; t < NT; ++t) { sB += Bc[t] * g[t]; sD += Dc[t] * g[t]; }
    out[O_MB2 + j] = pT * b2[j] + AT * Sb2[j] + sB;
    out[O_SB2 + j] = qT * Sb2[j] + sD;
  }
}

extern "C" void kernel_launch(void* const* d_in, const int* in_sizes, int n_in,
                              void* d_out, int out_size, void* d_ws, size_t ws_size,
                              hipStream_t stream) {
  (void)in_sizes; (void)n_in; (void)out_size; (void)ws_size;
  const float* x   = (const float*)d_in[0];
  const float* WK  = (const float*)d_in[1];
  const float* WV  = (const float*)d_in[2];
  const float* W1  = (const float*)d_in[3];
  const float* b1  = (const float*)d_in[4];
  const float* W2  = (const float*)d_in[5];
  const float* b2  = (const float*)d_in[6];
  const float* SW1 = (const float*)d_in[7];
  const float* Sb1 = (const float*)d_in[8];
  const float* SW2 = (const float*)d_in[9];
  const float* Sb2 = (const float*)d_in[10];
  float* out = (float*)d_out;
  float* ws  = (float*)d_ws;

  init_coeffs_kernel<<<1, 256, 0, stream>>>(ws, out);
  kv_kernel<<<dim3(NT / 16, NB, 2), 32, 0, stream>>>(x, WK, WV, ws);
  normalize_kernel<<<(2 * NB * NT * NE + 255) / 256, 256, 0, stream>>>(ws);
  forward_kernel<<<dim3(NT / 16, NB), 32, 0, stream>>>(W1, b1, W2, b2, ws, out);
  gradW1_kernel<<<dim3(NH / 16, NE / 16, NB), 32, 0, stream>>>(W1, SW1, ws, out);
  gradW2_kernel<<<dim3(NE / 16, NH / 16, NB), 32, 0, stream>>>(W2, SW2, ws, out);
  bias_kernel<<<(NB * NH + NB * NE + 255) / 256, 256, 0, stream>>>(b1, Sb1, b2, Sb2, ws, out);
}
